// OfflineReconstructorJetLatentSet2Set_38551626448944
// MI455X (gfx1250) — compile-verified, matrix operand order checked by reading, not run on hardware
//
#include <hip/hip_runtime.h>
#include <hip/hip_bf16.h>

// ---------------------------------------------------------------------------
// MI455X (gfx1250) implementation of the jet set2set transformer forward.
// All big GEMMs run on v_wmma_f32_16x16x32_f16 (f16 in, f32 accumulate),
// register-blocked 32x32 per wave (4 WMMAs/k-chunk) where M % 32 == 0.
// ---------------------------------------------------------------------------

static constexpr int Dm  = 256;   // model dim
static constexpr int Hh  = 8;     // heads
static constexpr int FFd = 1024;  // ffn dim
static constexpr int NLn = 8;     // layers
static constexpr int Gg  = 48;    // generator queries
static constexpr int DIN = 7;
static constexpr int PHh = 4;     // pool heads
static constexpr int Bb  = 32;
static constexpr int Ll  = 256;
static constexpr int CAND = Ll + Gg + 1; // 305

typedef __attribute__((ext_vector_type(16))) _Float16 v16h;
typedef __attribute__((ext_vector_type(8)))  float    v8f;

union HVec { v16h v; _Float16 h[16]; };
union FAcc { v8f  v; float    f[8];  };

__device__ __forceinline__ float dgelu(float x) {
  return 0.5f * x * (1.0f + erff(x * 0.70710678118654752440f));
}
__device__ __forceinline__ float dsigmoid(float x) { return 1.0f / (1.0f + expf(-x)); }
__device__ __forceinline__ float dsoftplus(float x) { return (x > 20.0f) ? x : log1pf(expf(x)); }
__device__ __forceinline__ float clampf(float v, float lo, float hi) { return fminf(fmaxf(v, lo), hi); }
__device__ __forceinline__ float wrap_phi(float t) { return atan2f(sinf(t), cosf(t)); }

__device__ __forceinline__ float blockReduceSum256(float v, float* sh) {
  int t = threadIdx.x;
  sh[t] = v; __syncthreads();
  for (int s = 128; s > 0; s >>= 1) { if (t < s) sh[t] += sh[t + s]; __syncthreads(); }
  float r = sh[0]; __syncthreads();
  return r;
}
__device__ __forceinline__ float blockReduceMax256(float v, float* sh) {
  int t = threadIdx.x;
  sh[t] = v; __syncthreads();
  for (int s = 128; s > 0; s >>= 1) { if (t < s) sh[t] = fmaxf(sh[t], sh[t + s]); __syncthreads(); }
  float r = sh[0]; __syncthreads();
  return r;
}

// ---------------------------------------------------------------------------
// Generic batched WMMA GEMM:  Y = act( alpha * X * op(W) + bias )
//   A(m,k) = X[m*ldx + k]
//   B(k,n) = transB ? W[n*ldw + k] : W[k*ldw + n]
//   Y(m,n) = Y[m*ldy + n]
// One wave computes a (16*MT)x32 output tile (MT A-fragments share the two
// B-fragments -> 2*MT WMMAs per 32-wide k-chunk).
// grid = (N/32, M/(16*MT), batch).
// Batch offset = (bz/inner)*outerStride + (bz%inner)*innerStride per tensor.
// ---------------------------------------------------------------------------
template <int MT>
__global__ void k_gemm_wmma(const float* __restrict__ Xb, const float* __restrict__ Wb,
                            const float* __restrict__ bias, float* __restrict__ Yb,
                            int K, long ldx, long ldw, long ldy,
                            int innerCount,
                            long xo, long xi, long wo_, long wi,
                            long yo, long yi,
                            int transB, int act, float alpha) {
  const int bz = blockIdx.z;
  const int ob = bz / innerCount;
  const int ib = bz - ob * innerCount;
  const float* X = Xb + (long)ob * xo + (long)ib * xi;
  const float* W = Wb + (long)ob * wo_ + (long)ib * wi;
  float*       Y = Yb + (long)ob * yo + (long)ib * yi;

  const int lane  = threadIdx.x;
  const int mrow  = lane & 15;
  const int ncol  = lane & 15;
  const int kbase = (lane >> 4) << 3;   // 0 for lanes 0-15, 8 for lanes 16-31
  const long m0 = (long)blockIdx.y * (16 * MT);
  const long n0 = (long)blockIdx.x * 32;

  const v8f vzero = {0.f,0.f,0.f,0.f,0.f,0.f,0.f,0.f};
  v8f acc0[MT], acc1[MT];
#pragma unroll
  for (int t = 0; t < MT; ++t) { acc0[t] = vzero; acc1[t] = vzero; }

  for (int kk = 0; kk < K; kk += 32) {
    HVec a[MT], b0, b1;
    if (transB) {
      const float* w0 = W + (n0 + ncol) * ldw + kk;
      const float* w1 = W + (n0 + 16 + ncol) * ldw + kk;
      if (kk + 32 < K) { __builtin_prefetch(w0 + 32, 0, 1); __builtin_prefetch(w1 + 32, 0, 1); }
#pragma unroll
      for (int i = 0; i < 16; ++i) {
        const int j = i >> 1, p = i & 1;
        const int k = (j < 4) ? (kbase + 2 * j + p) : (16 + kbase + 2 * (j - 4) + p);
        b0.h[i] = (_Float16)w0[k];
        b1.h[i] = (_Float16)w1[k];
      }
    } else {
      const float* wc = W + (long)kk * ldw;
#pragma unroll
      for (int i = 0; i < 16; ++i) {
        const int j = i >> 1, p = i & 1;
        const int k = (j < 4) ? (kbase + 2 * j + p) : (16 + kbase + 2 * (j - 4) + p);
        b0.h[i] = (_Float16)wc[(long)k * ldw + n0 + ncol];
        b1.h[i] = (_Float16)wc[(long)k * ldw + n0 + 16 + ncol];
      }
    }
#pragma unroll
    for (int t = 0; t < MT; ++t) {
      const float* xr = X + (m0 + 16 * t + mrow) * ldx + kk;
#pragma unroll
      for (int i = 0; i < 16; ++i) {
        const int j = i >> 1, p = i & 1;
        const int k = (j < 4) ? (kbase + 2 * j + p) : (16 + kbase + 2 * (j - 4) + p);
        a[t].h[i] = (_Float16)xr[k];
      }
    }
#pragma unroll
    for (int t = 0; t < MT; ++t) {
      acc0[t] = __builtin_amdgcn_wmma_f32_16x16x32_f16(false, a[t].v, false, b0.v, (short)0, acc0[t], false, false);
      acc1[t] = __builtin_amdgcn_wmma_f32_16x16x32_f16(false, a[t].v, false, b1.v, (short)0, acc1[t], false, false);
    }
  }

  const int mhi = (lane >> 4) << 3;
  const float bz0 = bias ? bias[n0 + ncol]      : 0.0f;
  const float bz1 = bias ? bias[n0 + 16 + ncol] : 0.0f;
#pragma unroll
  for (int t = 0; t < MT; ++t) {
    FAcc fa0, fa1; fa0.v = acc0[t]; fa1.v = acc1[t];
#pragma unroll
    for (int r = 0; r < 8; ++r) {
      const long m = m0 + 16 * t + mhi + r;
      float v0 = alpha * fa0.f[r] + bz0;
      float v1 = alpha * fa1.f[r] + bz1;
      if (act == 1) { v0 = dgelu(v0); v1 = dgelu(v1); }
      else if (act == 2) { v0 = dsigmoid(v0); v1 = dsigmoid(v1); }
      Y[m * ldy + n0 + ncol]      = v0;
      Y[m * ldy + n0 + 16 + ncol] = v1;
    }
  }
}

// ---------------------------------------------------------------------------
// Input embedding: x = gelu(LN(feat @ in_w^T + in_b))
// ---------------------------------------------------------------------------
__global__ void k_input_embed(const float* __restrict__ feat,
                              const float* __restrict__ in_w, const float* __restrict__ in_b,
                              const float* __restrict__ lg, const float* __restrict__ lb,
                              float* __restrict__ xout) {
  __shared__ float sh[256];
  const long row = blockIdx.x;
  const int  d   = threadIdx.x;
  const float* f = feat + row * DIN;
  float h = in_b[d];
#pragma unroll
  for (int i = 0; i < DIN; ++i) h += in_w[d * DIN + i] * f[i];
  float mean = blockReduceSum256(h, sh) * (1.0f / 256.0f);
  float var  = blockReduceSum256(h * h, sh) * (1.0f / 256.0f) - mean * mean;
  float y = (h - mean) * rsqrtf(var + 1e-5f) * lg[d] + lb[d];
  xout[row * Dm + d] = dgelu(y);
}

// ---------------------------------------------------------------------------
// Relative-position bias MLP (computed once, shared across all 8 layers)
//   bias[b,h,i,j] = rp_w2 @ gelu(rp_w1 @ [deta,dphi,dR] + rp_b1) + rp_b2
// ---------------------------------------------------------------------------
__global__ void k_rel_bias(const float* __restrict__ cst,
                           const float* __restrict__ w1, const float* __restrict__ b1,
                           const float* __restrict__ w2, const float* __restrict__ b2,
                           float* __restrict__ biasbuf) {
  __shared__ float sw1[64 * 3], sb1[64], sw2[8 * 64], sb2[8];
  const int t = threadIdx.x;
  if (t < 192) sw1[t] = w1[t];
  if (t < 64)  sb1[t] = b1[t];
  sw2[t]       = w2[t];
  sw2[t + 256] = w2[t + 256];
  if (t < 8) sb2[t] = b2[t];
  __syncthreads();

  const int i = blockIdx.x;
  const int b = blockIdx.y;
  const int j = t;
  const float* ci = cst + ((long)b * Ll + i) * 4;
  const float* cj = cst + ((long)b * Ll + j) * 4;
  const float deta = ci[1] - cj[1];
  const float dpr  = ci[2] - cj[2];
  const float dphi = atan2f(sinf(dpr), cosf(dpr));
  const float dR   = sqrtf(deta * deta + dphi * dphi + 1e-8f);

  float acc[8];
#pragma unroll
  for (int h = 0; h < 8; ++h) acc[h] = sb2[h];
  for (int k = 0; k < 64; ++k) {
    const float hk = dgelu(sw1[k * 3] * deta + sw1[k * 3 + 1] * dphi + sw1[k * 3 + 2] * dR + sb1[k]);
#pragma unroll
    for (int h = 0; h < 8; ++h) acc[h] += sw2[h * 64 + k] * hk;
  }
#pragma unroll
  for (int h = 0; h < 8; ++h)
    biasbuf[(((long)b * Hh + h) * Ll + i) * Ll + j] = acc[h];
}

// out = LN(a + (hasRes ? r : 0)) * g + be     (D = 256 per row)
__global__ void k_add_ln(const float* __restrict__ a, const float* __restrict__ r,
                         const float* __restrict__ g, const float* __restrict__ be,
                         float* __restrict__ out, int hasRes) {
  __shared__ float sh[256];
  const long row = blockIdx.x;
  const int  d   = threadIdx.x;
  float v = a[row * Dm + d];
  if (hasRes) v += r[row * Dm + d];
  float mean = blockReduceSum256(v, sh) * (1.0f / 256.0f);
  float var  = blockReduceSum256(v * v, sh) * (1.0f / 256.0f) - mean * mean;
  out[row * Dm + d] = (v - mean) * rsqrtf(var + 1e-5f) * g[d] + be[d];
}

// Masked softmax over last dim (256). mask_safe semantics: empty rows enable j==0.
__global__ void k_softmax(const float* __restrict__ s, const float* __restrict__ bias, int biasPerRow,
                          const unsigned char* __restrict__ mask, int rowsPerB, float negval,
                          float* __restrict__ outp) {
  __shared__ float sh[256];
  const long row = blockIdx.x;
  const int  b   = (int)(row / rowsPerB);
  const int  j   = threadIdx.x;
  const float mk = (mask[(long)b * Ll + j] != 0) ? 1.0f : 0.0f;
  const float anyv = blockReduceMax256(mk, sh);
  const bool on = (mk > 0.0f) || (j == 0 && anyv == 0.0f);
  float v = s[row * Ll + j];
  if (bias) v += biasPerRow ? bias[row * Ll + j] : bias[(long)b * Ll + j];
  v = on ? v : negval;
  const float mx = blockReduceMax256(v, sh);
  const float e  = expf(v - mx);
  const float sm = blockReduceSum256(e, sh);
  outp[row * Ll + j] = e / sm;
}

// y[n] = b[n] + W[n,:] @ x   (single vector)
__global__ void k_matvec(const float* __restrict__ W, const float* __restrict__ x,
                         const float* __restrict__ bv, float* __restrict__ y, int N, int K) {
  const int n = blockIdx.x * blockDim.x + threadIdx.x;
  if (n >= N) return;
  float acc = bv[n];
  for (int k = 0; k < K; ++k) acc += W[(long)n * K + k] * x[k];
  y[n] = acc;
}

// Small-N head: Y[m,n] = act(X[m,:] @ W[n,:] + b[n]),  N <= 4
__global__ void k_lin_small(const float* __restrict__ X, const float* __restrict__ W,
                            const float* __restrict__ bv, float* __restrict__ Y,
                            int M, int N, int K, int act) {
  const int m = blockIdx.x * blockDim.x + threadIdx.x;
  if (m >= M) return;
  float acc[4];
  for (int n = 0; n < N; ++n) acc[n] = bv[n];
  for (int k = 0; k < K; ++k) {
    const float xv = X[(long)m * K + k];
    for (int n = 0; n < N; ++n) acc[n] += W[(long)n * K + k] * xv;
  }
  for (int n = 0; n < N; ++n) {
    float v = acc[n];
    if (act == 2) v = dsigmoid(v);
    Y[(long)m * N + n] = v;
  }
}

__global__ void k_maskmul(float* __restrict__ w, const unsigned char* __restrict__ mask) {
  const long i = (long)blockIdx.x * blockDim.x + threadIdx.x;
  w[i] *= (mask[i] != 0) ? 1.0f : 0.0f;
}

// Pool attention scores (Lq = 1 per (b, ph), Dh = 64)
__global__ void k_pool_scores(const float* __restrict__ qpool, const float* __restrict__ kpool,
                              float* __restrict__ ps) {
  const int bp = blockIdx.x;
  const int b  = bp / PHh;
  const int ph = bp % PHh;
  const int j  = threadIdx.x;
  float acc = 0.0f;
  const float* kr = kpool + ((long)b * Ll + j) * Dm + ph * 64;
  const float* qr = qpool + ph * 64;
  for (int d = 0; d < 64; ++d) acc += qr[d] * kr[d];
  ps[(long)bp * Ll + j] = acc * 0.125f;   // 1/sqrt(64)
}

__global__ void k_pool_av(const float* __restrict__ ps, const float* __restrict__ vpool,
                          float* __restrict__ ctxpre) {
  const int b = blockIdx.x;
  const int d = threadIdx.x;
  const int ph = d >> 6;
  float acc = 0.0f;
  for (int j = 0; j < Ll; ++j)
    acc += ps[((long)b * PHh + ph) * Ll + j] * vpool[((long)b * Ll + j) * Dm + d];
  ctxpre[(long)b * Dm + d] = acc;
}

// Per-batch jet aggregates
__global__ void k_jet_stats(const float* __restrict__ cst, const unsigned char* __restrict__ mask,
                            float* __restrict__ jet) {
  __shared__ float sh[256];
  const int b = blockIdx.x;
  const int l = threadIdx.x;
  const float* c = cst + ((long)b * Ll + l) * 4;
  const float mv  = (mask[(long)b * Ll + l] != 0) ? 1.0f : 0.0f;
  const float pt  = fmaxf(c[0], 1e-8f);
  const float eta = clampf(c[1], -5.0f, 5.0f);
  const float phi = c[2];
  const float E   = fmaxf(c[3], 1e-8f);
  const float smv  = blockReduceSum256(mv, sh);
  const float spt  = blockReduceSum256(pt * mv, sh);
  const float sE   = blockReduceSum256(E * mv, sh);
  const float seta = blockReduceSum256(eta * mv, sh);
  const float ssin = blockReduceSum256(sinf(phi) * mv, sh);
  const float scos = blockReduceSum256(cosf(phi) * mv, sh);
  if (l == 0) {
    const float nv = fmaxf(smv, 1.0f);
    jet[b * 8 + 0] = logf(fmaxf(spt, 1e-8f));   // jet_logpt
    jet[b * 8 + 1] = logf(fmaxf(sE, 1e-8f));    // jet_logE
    jet[b * 8 + 2] = seta / nv;                 // jet_eta
    jet[b * 8 + 3] = atan2f(ssin, scos);        // jet_phi
  }
}

// Extra (generated) tokens -> cand_tokens[b, 256+g, :]
__global__ void k_extra(const float* __restrict__ graw, const float* __restrict__ jet,
                        float* __restrict__ cand_tok) {
  const int idx = blockIdx.x * blockDim.x + threadIdx.x;
  if (idx >= Bb * Gg) return;
  const int b = idx / Gg;
  const int g = idx % Gg;
  const float* gr = graw + (long)idx * 4;
  const float lp = jet[b * 8 + 0], le = jet[b * 8 + 1];
  const float je = jet[b * 8 + 2], jp = jet[b * 8 + 3];
  float ex_pt  = expf(clampf(lp + 0.9f * tanhf(gr[0]), -9.0f, 9.0f));
  float ex_eta = clampf(je + 0.85f * tanhf(gr[1]), -5.0f, 5.0f);
  float ex_phi = wrap_phi(jp + 0.85f * tanhf(gr[2]));
  float ex_E   = expf(clampf(le + 0.9f * tanhf(gr[3]), -9.0f, 11.0f));
  ex_E = fmaxf(ex_E, ex_pt * coshf(ex_eta));
  float* o = cand_tok + ((long)b * CAND + Ll + g) * 4;
  o[0] = ex_pt; o[1] = ex_eta; o[2] = ex_phi; o[3] = ex_E;
}

// Budget heads + weight rescaling (per batch)
__global__ void k_budget(const float* __restrict__ braw, const float* __restrict__ gex,
                         const float* __restrict__ tokw_raw,
                         float* __restrict__ out_extra_w,
                         float* __restrict__ out_btot, float* __restrict__ out_badd,
                         float* __restrict__ out_baux,
                         float* __restrict__ cand_w, float* __restrict__ cand_m,
                         float* __restrict__ cand_tok, float* __restrict__ wsScale) {
  __shared__ float sh[256];
  __shared__ float s_scaleExtra, s_btot, s_scaleTot;
  const int b = blockIdx.x;
  const int t = threadIdx.x;
  const float gx = (t < Gg) ? gex[(long)b * Gg + t] : 0.0f;
  const float sumex  = blockReduceSum256(gx, sh);
  const float sumtok = blockReduceSum256(tokw_raw[(long)b * Ll + t], sh);
  if (t == 0) {
    const float btot = dsoftplus(braw[b * 3 + 0]);
    const float badd = dsoftplus(braw[b * 3 + 1]);
    out_btot[b] = btot;
    out_badd[b] = badd;
    out_baux[b] = 0.0f;                        // budget_aux * 0.0
    s_scaleExtra = clampf(badd / fmaxf(sumex, 1e-8f), 0.25f, 4.0f);
    s_btot = btot;
  }
  __syncthreads();
  const float ew = (t < Gg) ? clampf(gx * s_scaleExtra, 0.0f, 1.0f) : 0.0f;
  const float sumew = blockReduceSum256(ew, sh);
  if (t == 0) s_scaleTot = clampf(s_btot / fmaxf(sumtok + sumew, 1e-8f), 0.25f, 4.0f);
  __syncthreads();
  if (t < Gg) {
    const float ewf = clampf(ew * s_scaleTot, 0.0f, 1.0f);
    out_extra_w[(long)b * Gg + t] = ewf;
    cand_w[(long)b * CAND + Ll + t] = ewf;
    cand_m[(long)b * CAND + Ll + t] = 1.0f;
  }
  if (t == 0) {
    wsScale[b] = s_scaleTot;
    cand_w[(long)b * CAND + (CAND - 1)] = 0.0f;
    cand_m[(long)b * CAND + (CAND - 1)] = 0.0f;
    float* dt = cand_tok + ((long)b * CAND + (CAND - 1)) * 4;
    dt[0] = 0.0f; dt[1] = 0.0f; dt[2] = 0.0f; dt[3] = 0.0f;
  }
}

// Final per-token outputs
__global__ void k_tokens(const float* __restrict__ cst,
                         const float* __restrict__ bd, const float* __restrict__ ba,
                         const float* __restrict__ tokw_raw, const float* __restrict__ wsScale,
                         float* __restrict__ cand_tok, float* __restrict__ cand_w,
                         float* __restrict__ cand_m, float* __restrict__ action,
                         float* __restrict__ out_tok_tokens, float* __restrict__ out_tokw) {
  const int idx = blockIdx.x * blockDim.x + threadIdx.x;
  if (idx >= Bb * Ll) return;
  const int b = idx / Ll;
  const int l = idx % Ll;
  const float* c = cst + (long)idx * 4;
  const float pt  = fmaxf(c[0], 1e-8f);
  const float eta = clampf(c[1], -5.0f, 5.0f);
  const float phi = c[2];
  const float E   = fmaxf(c[3], 1e-8f);
  const float* d = bd + (long)idx * 4;
  const float ba0 = 0.3f * tanhf(ba[(long)idx * 2 + 0]);
  const float ba1 = 0.3f * tanhf(ba[(long)idx * 2 + 1]);
  const float d_logpt = 0.7f * tanhf(d[0]);
  const float d_eta   = 0.5f * tanhf(d[1]) + ba0;
  const float d_phi   = 0.5f * tanhf(d[2]) + ba1;
  const float d_logE  = 0.7f * tanhf(d[3]);
  float tok_pt  = expf(clampf(logf(pt) + d_logpt, -9.0f, 9.0f));
  float tok_eta = clampf(eta + d_eta, -5.0f, 5.0f);
  float tok_phi = wrap_phi(phi + d_phi);
  float tok_E   = expf(clampf(logf(E) + d_logE, -9.0f, 11.0f));
  tok_E = fmaxf(tok_E, tok_pt * coshf(tok_eta));

  const float tw = clampf(tokw_raw[idx] * wsScale[b], 0.0f, 1.0f);

  float* tt = out_tok_tokens + (long)idx * 4;
  tt[0] = tok_pt; tt[1] = tok_eta; tt[2] = tok_phi; tt[3] = tok_E;
  float* ct = cand_tok + ((long)b * CAND + l) * 4;
  ct[0] = tok_pt; ct[1] = tok_eta; ct[2] = tok_phi; ct[3] = tok_E;
  cand_w[(long)b * CAND + l] = tw;
  cand_m[(long)b * CAND + l] = 0.0f;
  out_tokw[idx] = tw;
  float* ap = action + (long)idx * 4;
  ap[0] = tw / fmaxf(tw, 1e-8f);
  ap[1] = 0.0f; ap[2] = 0.0f; ap[3] = 0.0f;
}

// ---------------------------------------------------------------------------
// Host-side launch helper: pick 32-row register-blocked variant when possible
// ---------------------------------------------------------------------------
static inline void gemm(hipStream_t st,
                        const float* X, const float* W, const float* bias, float* Y,
                        int M, int N, int K,
                        long ldx, long ldw, long ldy,
                        int transB, int act, float alpha,
                        int batch = 1, int inner = 1,
                        long xo = 0, long xi = 0, long wo = 0, long wi = 0,
                        long yo = 0, long yi = 0) {
  if ((M & 31) == 0) {
    dim3 grid(N / 32, M / 32, batch);
    k_gemm_wmma<2><<<grid, 32, 0, st>>>(X, W, bias, Y, K, ldx, ldw, ldy,
                                        inner, xo, xi, wo, wi, yo, yi, transB, act, alpha);
  } else {
    dim3 grid(N / 32, M / 16, batch);
    k_gemm_wmma<1><<<grid, 32, 0, st>>>(X, W, bias, Y, K, ldx, ldw, ldy,
                                        inner, xo, xi, wo, wi, yo, yi, transB, act, alpha);
  }
}

extern "C" void kernel_launch(void* const* d_in, const int* in_sizes, int n_in,
                              void* d_out, int out_size, void* d_ws, size_t ws_size,
                              hipStream_t stream) {
  (void)in_sizes; (void)n_in; (void)out_size; (void)ws_size;
  const float*         feat = (const float*)d_in[0];
  const unsigned char* mask = (const unsigned char*)d_in[1];
  const float*         cst  = (const float*)d_in[2];
  auto P = [&](int i) { return (const float*)d_in[3 + i]; };
  // params (jax pytree / sorted-dict order):
  // 0 act_b  1 act_w  2 bud_b1 3 bud_b2 4 bud_w1 5 bud_w2 6 ge_b 7 ge_w
  // 8..15 gen_attn {bk,bo,bq,bv,wk,wo,wq,wv}
  // 16 gen_q 17 gh_b1 18 gh_b2 19 gh_w1 20 gh_w2 21 gn_b 22 gn_g
  // 23 in_b 24 in_ln_b 25 in_ln_g 26 in_w
  // 27+16*i layers {bk,bo,bq,bv,ff1_b,ff1_w,ff2_b,ff2_w,ln1_b,ln1_g,ln2_b,ln2_g,wk,wo,wq,wv}
  // 155..162 pool_attn {bk,bo,bq,bv,wk,wo,wq,wv}
  // 163 pool_q 164 ra_b1 165 ra_b2 166 ra_w1 167 ra_w2
  // 168 rp_b1 169 rp_b2 170 rp_w1 171 rp_w2 172 sd_b 173 sd_w
  // 174 se_b 175 se_w 176 tn_b 177 tn_g 178 un_b1 179 un_b2 180 un_w1 181 un_w2

  float* ws = (float*)d_ws;
  size_t o = 0;
  float* x      = ws + o; o += (size_t)Bb * Ll * Dm;       // 2M
  float* t1     = ws + o; o += (size_t)Bb * Ll * Dm;
  float* qb     = ws + o; o += (size_t)Bb * Ll * Dm;
  float* kb     = ws + o; o += (size_t)Bb * Ll * Dm;
  float* vb     = ws + o; o += (size_t)Bb * Ll * Dm;
  float* ao     = ws + o; o += (size_t)Bb * Ll * Dm;
  float* ffb    = ws + o; o += (size_t)Bb * Ll * FFd;      // 8M
  float* biasb  = ws + o; o += (size_t)Bb * Hh * Ll * Ll;  // 16M
  float* sc     = ws + o; o += (size_t)Bb * Hh * Ll * Ll;  // 16M (reused)
  float* sc2    = sc + (size_t)8 * 1024 * 1024;            // assign logits (inside sc region)
  float* qpool  = ws + o; o += 256;
  float* ctxpre = ws + o; o += (size_t)Bb * Dm;
  float* ctx    = ws + o; o += (size_t)Bb * Dm;
  float* budh   = ws + o; o += (size_t)Bb * Dm;
  float* brawb  = ws + o; o += 128;
  float* qg     = ws + o; o += (size_t)Gg * Dm;
  float* gatt   = ws + o; o += (size_t)Bb * Gg * Dm;
  float* gdec   = ws + o; o += (size_t)Bb * Gg * Dm;
  float* ghb    = ws + o; o += (size_t)Bb * Gg * Dm;
  float* graw   = ws + o; o += (size_t)Bb * Gg * 4;
  float* gex    = ws + o; o += (size_t)Bb * Gg;
  float* pex    = ws + o; o += (size_t)Bb * Gg * Dm;
  float* bdb    = ws + o; o += (size_t)Bb * Ll * 4;
  float* bab    = ws + o; o += (size_t)Bb * Ll * 2;
  float* tokw   = ws + o; o += (size_t)Bb * Ll;
  float* prior  = ws + o; o += (size_t)Bb * Ll;
  float* jet    = ws + o; o += (size_t)Bb * 8;
  float* wsScale= ws + o; o += 32;

  // output regions (fp32, flat concat in return order)
  float* out        = (float*)d_out;
  float* o_cand_tok = out;                                   // (B,305,4)
  float* o_cand_w   = o_cand_tok + (size_t)Bb * CAND * 4;    // (B,305)
  float* o_cand_m   = o_cand_w   + (size_t)Bb * CAND;        // (B,305)
  float* o_action   = o_cand_m   + (size_t)Bb * CAND;        // (B,L,4)
  float* o_extra_w  = o_action   + (size_t)Bb * Ll * 4;      // (B,G)
  float* o_btot     = o_extra_w  + (size_t)Bb * Gg;          // (B,)
  float* o_badd     = o_btot     + Bb;                       // (B,)
  float* o_baux     = o_badd     + Bb;                       // (B,)
  float* o_tok_tok  = o_baux     + Bb;                       // (B,L,4)
  float* o_tokw     = o_tok_tok  + (size_t)Bb * Ll * 4;      // (B,L)
  float* o_e2b      = o_tokw     + (size_t)Bb * Ll;          // (B,G,L)

  const long LD = (long)Ll * Dm;   // per-batch stride in x/q/k/v
  const float inv_sqrt_dh = 0.17677669529663688f;  // 1/sqrt(32)
  const int BL = Bb * Ll;

  // ---- input embedding + relative position bias (once) ----
  k_input_embed<<<BL, 256, 0, stream>>>(feat, P(26), P(23), P(25), P(24), x);
  k_rel_bias<<<dim3(Ll, Bb), 256, 0, stream>>>(cst, P(170), P(168), P(171), P(169), biasb);

  // ---- encoder layers ----
  for (int li = 0; li < NLn; ++li) {
    const int pb = 27 + 16 * li;
    const float *lbk = P(pb+0), *lbo = P(pb+1), *lbq = P(pb+2), *lbv = P(pb+3);
    const float *f1b = P(pb+4), *f1w = P(pb+5), *f2b = P(pb+6), *f2w = P(pb+7);
    const float *l1b = P(pb+8), *l1g = P(pb+9), *l2b = P(pb+10), *l2g = P(pb+11);
    const float *lwk = P(pb+12), *lwo = P(pb+13), *lwq = P(pb+14), *lwv = P(pb+15);

    gemm(stream, x, lwq, lbq, qb, BL, Dm, Dm, Dm, Dm, Dm, 1, 0, 1.0f);
    gemm(stream, x, lwk, lbk, kb, BL, Dm, Dm, Dm, Dm, Dm, 1, 0, 1.0f);
    gemm(stream, x, lwv, lbv, vb, BL, Dm, Dm, Dm, Dm, Dm, 1, 0, 1.0f);

    // scores[b,h] = Q K^T / sqrt(32)    (batched over B*H)
    gemm(stream, qb, kb, nullptr, sc, Ll, Ll, 32, Dm, Dm, Ll, 1, 0, inv_sqrt_dh,
         Bb * Hh, Hh, LD, 32, LD, 32, (long)Hh * Ll * Ll, (long)Ll * Ll);
    k_softmax<<<Bb * Hh * Ll, 256, 0, stream>>>(sc, biasb, 1, mask, Hh * Ll, -1e9f, sc);
    // O[b,h] = A V
    gemm(stream, sc, vb, nullptr, ao, Ll, 32, Ll, Ll, Dm, Dm, 0, 0, 1.0f,
         Bb * Hh, Hh, (long)Hh * Ll * Ll, (long)Ll * Ll, LD, 32, LD, 32);
    gemm(stream, ao, lwo, lbo, t1, BL, Dm, Dm, Dm, Dm, Dm, 1, 0, 1.0f);
    k_add_ln<<<BL, 256, 0, stream>>>(t1, x, l1g, l1b, x, 1);

    gemm(stream, x, f1w, f1b, ffb, BL, FFd, Dm, Dm, Dm, FFd, 1, 1, 1.0f);
    gemm(stream, ffb, f2w, f2b, t1, BL, Dm, FFd, FFd, FFd, Dm, 1, 0, 1.0f);
    k_add_ln<<<BL, 256, 0, stream>>>(t1, x, l2g, l2b, x, 1);
  }

  // ---- final trunk norm ----
  k_add_ln<<<BL, 256, 0, stream>>>(x, nullptr, P(177), P(176), x, 0);

  // ---- per-token heads ----
  gemm(stream, x, P(180), P(178), t1, BL, Dm, Dm, Dm, Dm, Dm, 1, 1, 1.0f);      // un_w1 + gelu
  k_lin_small<<<BL / 256, 256, 0, stream>>>(t1, P(181), P(179), bdb, BL, 4, Dm, 0);
  gemm(stream, x, P(166), P(164), ffb, BL, 128, Dm, Dm, Dm, 128, 1, 1, 1.0f);   // ra_w1 + gelu
  k_lin_small<<<BL / 256, 256, 0, stream>>>(ffb, P(167), P(165), bab, BL, 2, 128, 0);
  k_lin_small<<<BL / 256, 256, 0, stream>>>(x, P(1), P(0), tokw, BL, 1, Dm, 2); // sigmoid
  k_maskmul<<<BL / 256, 256, 0, stream>>>(tokw, mask);
  k_lin_small<<<BL / 256, 256, 0, stream>>>(x, P(175), P(174), prior, BL, 1, Dm, 0);

  // ---- pooled context attention (PH=4, Dh=64, Lq=1) ----
  gemm(stream, x, P(159), P(155), kb, BL, Dm, Dm, Dm, Dm, Dm, 1, 0, 1.0f);      // pool K
  gemm(stream, x, P(162), P(158), vb, BL, Dm, Dm, Dm, Dm, Dm, 1, 0, 1.0f);      // pool V
  k_matvec<<<1, 256, 0, stream>>>(P(161), P(163), P(157), qpool, Dm, Dm);       // pool Q vec
  k_pool_scores<<<Bb * PHh, 256, 0, stream>>>(qpool, kb, sc);
  k_softmax<<<Bb * PHh, 256, 0, stream>>>(sc, nullptr, 0, mask, PHh, -1e9f, sc);
  k_pool_av<<<Bb, 256, 0, stream>>>(sc, vb, ctxpre);
  gemm(stream, ctxpre, P(160), P(156), ctx, Bb, Dm, Dm, Dm, Dm, Dm, 1, 0, 1.0f);

  // ---- budget head ----
  gemm(stream, ctx, P(4), P(2), budh, Bb, Dm, Dm, Dm, Dm, Dm, 1, 1, 1.0f);
  k_lin_small<<<1, 256, 0, stream>>>(budh, P(5), P(3), brawb, Bb, 3, Dm, 0);

  // ---- generator attention (H=8, Dh=32, G=48 queries, shared across batch) ----
  gemm(stream, P(16), P(14), P(10), qg, Gg, Dm, Dm, Dm, Dm, Dm, 1, 0, 1.0f);    // gen Q
  gemm(stream, x, P(12), P(8),  kb, BL, Dm, Dm, Dm, Dm, Dm, 1, 0, 1.0f);        // gen K
  gemm(stream, x, P(15), P(11), vb, BL, Dm, Dm, Dm, Dm, Dm, 1, 0, 1.0f);        // gen V
  gemm(stream, qg, kb, nullptr, sc, Gg, Ll, 32, Dm, Dm, Ll, 1, 0, inv_sqrt_dh,
       Bb * Hh, Hh, 0, 32, LD, 32, (long)Hh * Gg * Ll, (long)Gg * Ll);
  k_softmax<<<Bb * Hh * Gg, 256, 0, stream>>>(sc, nullptr, 0, mask, Hh * Gg, -1e9f, sc);
  gemm(stream, sc, vb, nullptr, gatt, Gg, 32, Ll, Ll, Dm, Dm, 0, 0, 1.0f,
       Bb * Hh, Hh, (long)Hh * Gg * Ll, (long)Gg * Ll, LD, 32, (long)Gg * Dm, 32);
  gemm(stream, gatt, P(13), P(9), t1, Bb * Gg, Dm, Dm, Dm, Dm, Dm, 1, 0, 1.0f); // gen O proj
  k_add_ln<<<Bb * Gg, 256, 0, stream>>>(t1, nullptr, P(22), P(21), gdec, 0);    // gn LN

  gemm(stream, gdec, P(19), P(17), ghb, Bb * Gg, Dm, Dm, Dm, Dm, Dm, 1, 1, 1.0f);
  k_lin_small<<<Bb * Gg / 256, 256, 0, stream>>>(ghb, P(20), P(18), graw, Bb * Gg, 4, Dm, 0);
  k_lin_small<<<Bb * Gg / 256, 256, 0, stream>>>(gdec, P(7), P(6), gex, Bb * Gg, 1, Dm, 2);
  gemm(stream, gdec, P(173), P(172), pex, Bb * Gg, Dm, Dm, Dm, Dm, Dm, 1, 0, 1.0f);

  // ---- assignment logits + softmax -> extra_to_base output ----
  gemm(stream, pex, x, nullptr, sc2, Gg, Ll, Dm, Dm, Dm, Ll, 1, 0, 1.0f / 16.0f,
       Bb, 1, (long)Gg * Dm, 0, LD, 0, (long)Gg * Ll, 0);
  k_softmax<<<Bb * Gg, 256, 0, stream>>>(sc2, prior, 0, mask, Gg, -1e4f, o_e2b);

  // ---- final outputs ----
  k_jet_stats<<<Bb, 256, 0, stream>>>(cst, mask, jet);
  k_extra<<<(Bb * Gg + 255) / 256, 256, 0, stream>>>(graw, jet, o_cand_tok);
  k_budget<<<Bb, 256, 0, stream>>>(brawb, gex, tokw, o_extra_w, o_btot, o_badd, o_baux,
                                   o_cand_w, o_cand_m, o_cand_tok, wsScale);
  k_tokens<<<BL / 256, 256, 0, stream>>>(cst, bdb, bab, tokw, wsScale,
                                         o_cand_tok, o_cand_w, o_cand_m, o_action,
                                         o_tok_tok, o_tokw);
}
